// BertWithCRF_72567767433797
// MI455X (gfx1250) — compile-verified, matrix operand order checked by reading, not run on hardware
//
#include <hip/hip_runtime.h>
#include <hip/hip_bf16.h>
#include <math.h>

#define Bq   8
#define Sq   512
#define Hq   768
#define Lq   12
#define NHq  12
#define DHq  64
#define DFFq 3072
#define NLq  19
#define TOK  (Bq*Sq)   // 4096 token rows

typedef __attribute__((ext_vector_type(16))) __bf16 v16bf;
typedef __attribute__((ext_vector_type(8)))  __bf16 bf16x8;
typedef __attribute__((ext_vector_type(8)))  float  v8f;
typedef __attribute__((ext_vector_type(4)))  float  f32x4;
typedef __attribute__((ext_vector_type(4)))  unsigned u32x4;
typedef __attribute__((ext_vector_type(8)))  int      i32x8;
typedef __attribute__((ext_vector_type(4)))  int      i32x4;

// native f32 -> bf16 convert (backend emits v_cvt*bf16* on gfx1250)
static __device__ __forceinline__ __bf16 f2bf(float f){ return (__bf16)f; }

static __device__ __forceinline__ v16bf pack16(bf16x8 lo, bf16x8 hi){
  return __builtin_shufflevector(lo,hi,0,1,2,3,4,5,6,7,8,9,10,11,12,13,14,15);
}

static __device__ __forceinline__ v8f wmma_bf16(v16bf a, v16bf b, v8f c){
  return __builtin_amdgcn_wmma_f32_16x16x32_bf16(false, a, false, b, (short)0, c, false, false);
}

// ---- Tensor Data Mover: 2-D f32 tile (tile_x contiguous elems, tile_y rows) ------
// Descriptor bitfields per CDNA5 ISA sec 8.3/8.4 (groups 2/3 zero => 2-D tensor).
// clang-23 builtin form: (u32x4 g0, i32x8 g1, i32x4 g2, i32x4 g3, i32x8, i32 cpol)
static __device__ __forceinline__ void tdm_load_2d_f32(unsigned lds_addr, const void* gaddr,
                                                       unsigned tile_x, unsigned tile_y,
                                                       unsigned tensor_x, unsigned tensor_y,
                                                       unsigned long long stride_x){
  unsigned long long ga = (unsigned long long)(uintptr_t)gaddr;
  u32x4 g0;
  g0[0] = 1u;                                       // count=1, user mode
  g0[1] = lds_addr;                                 // lds_addr [63:32]
  g0[2] = (unsigned)(ga & 0xffffffffu);             // global_addr [95:64]
  g0[3] = (unsigned)((ga >> 32) & 0x01ffffffu)      // global_addr [120:96]
        | (2u << 30);                               // type=2 ("image")
  i32x8 g1;
  g1[0] = (int)(2u << 16);                          // data_size=2 (4 bytes), mask=0
  g1[1] = (int)((tensor_x & 0xffffu) << 16);        // tensor_dim0 [79:48] lo
  g1[2] = (int)((tensor_x >> 16) | ((tensor_y & 0xffffu) << 16));
  g1[3] = (int)((tensor_y >> 16) | ((tile_x & 0xffffu) << 16)); // tile_dim0 [127:112]
  g1[4] = (int)(tile_y & 0xffffu);                  // tile_dim1 [143:128]
  g1[5] = (int)(stride_x & 0xffffffffu);            // tensor_dim0_stride [207:160] lo
  g1[6] = (int)((stride_x >> 32) & 0xffffu);        //   ... hi 16
  g1[7] = 0;
  i32x4 z4 = {0,0,0,0};
  i32x8 z8 = {0,0,0,0,0,0,0,0};
  __builtin_amdgcn_tensor_load_to_lds(g0, g1, z4, z4, z8, 0);
}

// 16x32 bf16 A/B fragment from row-major f32 (row stride `stride`), base at (row0,k0).
// ISA layout: lane m = L&15, ks = (L>>4)*8, halves at k=ks..ks+7 and ks+16..ks+23.
static __device__ __forceinline__ v16bf frag_row_f32(const float* base, int stride, int lane){
  const float* p = base + (size_t)(lane&15)*stride + ((lane>>4)<<3);
  f32x4 x0=*(const f32x4*)(p),    x1=*(const f32x4*)(p+4);
  f32x4 x2=*(const f32x4*)(p+16), x3=*(const f32x4*)(p+20);
  v16bf f;
  #pragma unroll
  for(int j=0;j<4;++j){ f[j]=f2bf(x0[j]); f[4+j]=f2bf(x1[j]); f[8+j]=f2bf(x2[j]); f[12+j]=f2bf(x3[j]); }
  return f;
}

// B fragment where lane picks column n (stride-1 dim) and k advances by `stride` rows.
static __device__ __forceinline__ v16bf frag_col_f32(const float* base, int stride, int lane){
  const float* p = base + (lane&15) + (size_t)((lane>>4)<<3)*stride;
  v16bf f;
  #pragma unroll
  for(int j=0;j<8;++j){ f[j]=f2bf(p[(size_t)j*stride]); f[8+j]=f2bf(p[(size_t)(16+j)*stride]); }
  return f;
}

// ---------------- GEMM: C[M,N] = A[M,K] @ W[K,N] + bias, optional GELU -----------
// A tile (128x32 f32) staged by the Tensor Data Mover; B tile staged as transposed
// bf16 so B-fragments are contiguous 16B LDS loads.
__global__ __launch_bounds__(256) void gemm_kernel(const float* __restrict__ A, const float* __restrict__ W,
                                                   const float* __restrict__ bias, float* __restrict__ C,
                                                   int M, int N, int K, int fuse){
  __shared__ float  Af[128][32];   // 16 KB, filled by TDM
  __shared__ __bf16 BsT[64][32];   // 4 KB, stored N-major
  const int tid = threadIdx.x, lane = tid & 31, wave = tid >> 5;
  const int bm = blockIdx.y*128, bn = blockIdx.x*64;
  const int wm = (wave & 3)*32, wn = (wave >> 2)*32;
  const unsigned ldsA = (unsigned)(uintptr_t)&Af[0][0];
  v8f c[2][2] = {};
  for(int k0=0;k0<K;k0+=32){
    if (tid < 32) {  // wave 0 owns the TDM issue (EXEC ignored by tensor ops)
      tdm_load_2d_f32(ldsA, A + (size_t)bm*K + k0,
                      /*tile_x=*/32, /*tile_y=*/128,
                      /*tensor_x=*/(unsigned)K, /*tensor_y=*/(unsigned)M,
                      /*stride_x=*/(unsigned long long)K);
    }
    { // stage B tile 32x64 -> transposed into BsT[n][k], f32->bf16 in regs
      int kk = tid>>3, nb = (tid&7)*8;
      const float* src = W + (size_t)(k0+kk)*N + bn + nb;
      if(k0+32<K) __builtin_prefetch(W + (size_t)(k0+32+kk)*N + bn + nb, 0, 1);
      f32x4 y0=*(const f32x4*)src, y1=*(const f32x4*)(src+4);
      #pragma unroll
      for(int j=0;j<4;++j){ BsT[nb+j][kk]=f2bf(y0[j]); BsT[nb+4+j][kk]=f2bf(y1[j]); }
    }
    if (tid < 32) __builtin_amdgcn_s_wait_tensorcnt(0);  // issuing wave drains TDM
    __syncthreads();                                     // publish Af + BsT
    v16bf a[2], b[2];
    #pragma unroll
    for(int i=0;i<2;++i){  // A frag from f32 LDS, convert in regs
      const float* p = &Af[wm + i*16 + (lane&15)][(lane>>4)<<3];
      f32x4 x0=*(const f32x4*)p,      x1=*(const f32x4*)(p+4);
      f32x4 x2=*(const f32x4*)(p+16), x3=*(const f32x4*)(p+20);
      v16bf f;
      #pragma unroll
      for(int j=0;j<4;++j){ f[j]=f2bf(x0[j]); f[4+j]=f2bf(x1[j]); f[8+j]=f2bf(x2[j]); f[12+j]=f2bf(x3[j]); }
      a[i] = f;
    }
    #pragma unroll
    for(int j=0;j<2;++j){
      const __bf16* p = &BsT[wn + j*16 + (lane&15)][(lane>>4)<<3];
      b[j] = pack16(*(const bf16x8*)p, *(const bf16x8*)(p+16));
    }
    #pragma unroll
    for(int i=0;i<2;++i)
      #pragma unroll
      for(int j=0;j<2;++j)
        c[i][j] = wmma_bf16(a[i], b[j], c[i][j]);
    __syncthreads();
  }
  // epilogue: C/D layout m = r + 8*(lane>=16), n = lane&15
  #pragma unroll
  for(int i=0;i<2;++i){
    #pragma unroll
    for(int j=0;j<2;++j){
      int n = bn + wn + j*16 + (lane&15);
      float bv = bias ? bias[n] : 0.f;
      #pragma unroll
      for(int r=0;r<8;++r){
        int m = bm + wm + i*16 + r + 8*(lane>>4);
        float val = c[i][j][r] + bv;
        if(fuse==1){ float u=val; val = 0.5f*u*(1.f + tanhf(0.7978845608f*(u + 0.044715f*u*u*u))); }
        C[(size_t)m*N + n] = val;
      }
    }
  }
}

// ---------------- Attention: per-wave (b, head, 16-query tile) --------------------
__global__ __launch_bounds__(64) void attn_kernel(const float* __restrict__ Q, const float* __restrict__ K,
                                                  const float* __restrict__ V, const int* __restrict__ amask,
                                                  float* __restrict__ O){
  extern __shared__ char smem[];                  // 2*(32KB scores f32) + 2*(16KB probs bf16)
  const int lane = threadIdx.x & 31;
  const int wave = threadIdx.x >> 5;
  float*  sc = (float*)(smem + (size_t)wave*16*512*4);
  __bf16* pb = (__bf16*)(smem + 65536 + (size_t)wave*16*512*2);
  const int task = blockIdx.x*2 + wave;           // B*NH*(S/16) = 3072 tasks
  const int qt = task & 31;
  const int hh = (task >> 5) % NHq;
  const int bb = task / (32*NHq);
  const float* Qb = Q + ((size_t)bb*Sq + qt*16)*Hq + hh*DHq;
  const float* Kb = K + (size_t)bb*Sq*Hq + hh*DHq;
  const float* Vb = V + (size_t)bb*Sq*Hq + hh*DHq;
  const float scale = 0.125f;                     // 1/sqrt(64)

  // scores = Q @ K^T : K rows are contiguous in d, so B-frags are row-frags of K
  v16bf aq0 = frag_row_f32(Qb,      Hq, lane);    // d = 0..31
  v16bf aq1 = frag_row_f32(Qb + 32, Hq, lane);    // d = 32..63
  for(int n0=0;n0<Sq;n0+=16){
    v16bf bk0 = frag_row_f32(Kb + (size_t)n0*Hq,      Hq, lane);
    v16bf bk1 = frag_row_f32(Kb + (size_t)n0*Hq + 32, Hq, lane);
    v8f c = {};
    c = wmma_bf16(aq0, bk0, c);
    c = wmma_bf16(aq1, bk1, c);
    int n = n0 + (lane&15);
    float biasv = (1.f - (float)amask[bb*Sq + n]) * -1e9f;
    int mrow = 8*(lane>>4);
    #pragma unroll
    for(int r=0;r<8;++r) sc[(size_t)(mrow+r)*512 + n] = c[r]*scale + biasv;
  }
  __syncthreads();
  // softmax per row, write bf16 probs
  if(lane < 16){
    const float* row = sc + (size_t)lane*512;
    float mx = -3.4e38f;
    for(int i=0;i<512;++i) mx = fmaxf(mx, row[i]);
    float ssum = 0.f;
    for(int i=0;i<512;++i) ssum += __expf(row[i]-mx);
    float inv = 1.f/ssum;
    __bf16* prow = pb + (size_t)lane*512;
    for(int i=0;i<512;++i) prow[i] = f2bf(__expf(row[i]-mx)*inv);
  }
  __syncthreads();
  // ctx = probs @ V : A-frags straight from LDS bf16, B-frags are V columns (stride H)
  v8f o[4] = {};
  for(int k0=0;k0<Sq;k0+=32){
    const __bf16* pp = pb + (size_t)(lane&15)*512 + k0 + ((lane>>4)<<3);
    v16bf a = pack16(*(const bf16x8*)pp, *(const bf16x8*)(pp+16));
    #pragma unroll
    for(int j=0;j<4;++j){
      v16bf bcol = frag_col_f32(Vb + (size_t)k0*Hq + j*16, Hq, lane);
      o[j] = wmma_bf16(a, bcol, o[j]);
    }
  }
  float* Ob = O + ((size_t)bb*Sq + qt*16)*Hq + hh*DHq;
  int mrow = 8*(lane>>4);
  #pragma unroll
  for(int j=0;j<4;++j){
    int n = j*16 + (lane&15);
    #pragma unroll
    for(int r=0;r<8;++r) Ob[(size_t)(mrow+r)*Hq + n] = o[j][r];
  }
}

// ---------------- Embedding gather ------------------------------------------------
__global__ __launch_bounds__(256) void embed_kernel(const int* __restrict__ ids, const float* __restrict__ tok,
                                                    const float* __restrict__ pos, float* __restrict__ out){
  const int t = blockIdx.x;
  const int s = t & (Sq-1);
  const int id = ids[t];
  const float* tr = tok + (size_t)id*Hq;
  const float* pr = pos + (size_t)s*Hq;
  float* orow = out + (size_t)t*Hq;
  for(int i=threadIdx.x;i<Hq;i+=256) orow[i] = tr[i] + pr[i];
}

// ---------------- Fused residual + LayerNorm (one block per token row) ------------
__global__ __launch_bounds__(256) void add_ln_kernel(float* __restrict__ out, const float* __restrict__ a,
                                                     const float* __restrict__ bsrc, const float* __restrict__ gamma,
                                                     const float* __restrict__ beta){
  __shared__ float buf[Hq];
  __shared__ float rs[8], rq[8], stats[2];
  const int row = blockIdx.x, tid = threadIdx.x, lane = tid & 31, wv = tid >> 5;
  float s = 0.f, s2 = 0.f;
  for(int i=tid;i<Hq;i+=256){
    float v = a[(size_t)row*Hq + i] + (bsrc ? bsrc[(size_t)row*Hq + i] : 0.f);
    buf[i] = v; s += v; s2 += v*v;
  }
  #pragma unroll
  for(int o=16;o>0;o>>=1){ s += __shfl_down(s,o,32); s2 += __shfl_down(s2,o,32); }
  if(lane==0){ rs[wv]=s; rq[wv]=s2; }
  __syncthreads();
  if(tid==0){
    float ts=0.f,t2=0.f;
    for(int w=0;w<8;++w){ ts+=rs[w]; t2+=rq[w]; }
    float mean = ts/(float)Hq;
    float var  = t2/(float)Hq - mean*mean;
    stats[0]=mean; stats[1]=rsqrtf(var + 1e-12f);
  }
  __syncthreads();
  float mean=stats[0], rstd=stats[1];
  for(int i=tid;i<Hq;i+=256)
    out[(size_t)row*Hq + i] = (buf[i]-mean)*rstd*gamma[i] + beta[i];
}

// ---------------- Classifier (N=19, skinny => plain FMA, memory bound) -----------
__global__ __launch_bounds__(256) void cls_kernel(const float* __restrict__ x, const float* __restrict__ Wc,
                                                  const float* __restrict__ bc, float* __restrict__ logits){
  const int t = blockIdx.x*256 + threadIdx.x;
  const float* xr = x + (size_t)t*Hq;
  float accv[NLq];
  #pragma unroll
  for(int j=0;j<NLq;++j) accv[j]=bc[j];
  for(int i=0;i<Hq;++i){
    float xv = xr[i];
    #pragma unroll
    for(int j=0;j<NLq;++j) accv[j] += xv * Wc[i*NLq + j];
  }
  for(int j=0;j<NLq;++j) logits[(size_t)t*NLq + j] = accv[j];
}

// ---------------- CRF: numerator + forward-algorithm denominator -> scalar loss ---
__global__ __launch_bounds__(256) void crf_kernel(const float* __restrict__ logits, const int* __restrict__ labels,
                                                  const int* __restrict__ amask, const float* __restrict__ startT,
                                                  const float* __restrict__ endT, const float* __restrict__ trans,
                                                  float* __restrict__ out){
  __shared__ float alpha[Bq][NLq];
  __shared__ float acc[Bq];
  const int tid = threadIdx.x;
  const int b = tid / NLq, j = tid % NLq;
  const bool act = tid < Bq*NLq;
  if(act) alpha[b][j] = startT[j] + logits[(size_t)b*Sq*NLq + j];
  __syncthreads();
  for(int t=1;t<Sq;++t){
    float v = 0.f;
    if(act){
      float mx = -3.4e38f;
      #pragma unroll
      for(int i=0;i<NLq;++i) mx = fmaxf(mx, alpha[b][i] + trans[i*NLq + j]);
      float s = 0.f;
      #pragma unroll
      for(int i=0;i<NLq;++i) s += __expf(alpha[b][i] + trans[i*NLq + j] - mx);
      v = mx + __logf(s) + logits[((size_t)b*Sq + t)*NLq + j];
      if(amask[b*Sq + t] <= 0) v = alpha[b][j];
    }
    __syncthreads();
    if(act) alpha[b][j] = v;
    __syncthreads();
  }
  if(tid < Bq){
    int bb = tid;
    float mx = -3.4e38f;
    for(int i=0;i<NLq;++i) mx = fmaxf(mx, alpha[bb][i] + endT[i]);
    float s = 0.f;
    for(int i=0;i<NLq;++i) s += __expf(alpha[bb][i] + endT[i] - mx);
    float den = mx + __logf(s);
    int l0 = labels[bb*Sq];
    float num = startT[l0] + logits[(size_t)bb*Sq*NLq + l0];
    int prev = l0;
    int msum = (amask[bb*Sq] > 0) ? 1 : 0;
    for(int t=1;t<Sq;++t){
      int lt = labels[bb*Sq + t];
      if(amask[bb*Sq + t] > 0){
        num += trans[prev*NLq + lt] + logits[((size_t)bb*Sq + t)*NLq + lt];
        msum += 1;
      }
      prev = lt;
    }
    int lastTag = labels[bb*Sq + (msum - 1)];
    num += endT[lastTag];
    acc[bb] = den - num;
  }
  __syncthreads();
  if(tid == 0){
    float s = 0.f;
    for(int b2=0;b2<Bq;++b2) s += acc[b2];
    out[0] = s;
  }
}

extern "C" void kernel_launch(void* const* d_in, const int* in_sizes, int n_in,
                              void* d_out, int out_size, void* d_ws, size_t ws_size,
                              hipStream_t stream){
  (void)in_sizes; (void)n_in; (void)out_size; (void)ws_size;
  const int*   input_ids = (const int*)  d_in[0];
  const int*   amask     = (const int*)  d_in[1];
  const int*   labels    = (const int*)  d_in[2];
  const float* token_emb = (const float*)d_in[3];
  const float* pos_emb   = (const float*)d_in[4];
  const float* ln_emb_s  = (const float*)d_in[5];
  const float* ln_emb_b  = (const float*)d_in[6];
  const float* Wq  = (const float*)d_in[7];
  const float* bqv = (const float*)d_in[8];
  const float* Wk  = (const float*)d_in[9];
  const float* bkv = (const float*)d_in[10];
  const float* Wv  = (const float*)d_in[11];
  const float* bvv = (const float*)d_in[12];
  const float* Wo  = (const float*)d_in[13];
  const float* bov = (const float*)d_in[14];
  const float* ln1_s = (const float*)d_in[15];
  const float* ln1_b = (const float*)d_in[16];
  const float* W1  = (const float*)d_in[17];
  const float* b1v = (const float*)d_in[18];
  const float* W2  = (const float*)d_in[19];
  const float* b2v = (const float*)d_in[20];
  const float* ln2_s = (const float*)d_in[21];
  const float* ln2_b = (const float*)d_in[22];
  const float* cls_W = (const float*)d_in[23];
  const float* cls_b = (const float*)d_in[24];
  const float* startT = (const float*)d_in[25];
  const float* endT   = (const float*)d_in[26];
  const float* trans  = (const float*)d_in[27];

  char* ws = (char*)d_ws;
  size_t off = 0;
  auto take = [&](size_t elems)->float*{
    float* p = (float*)(ws + off);
    off += (elems*sizeof(float) + 255) & ~(size_t)255;
    return p;
  };
  float* x      = take((size_t)TOK*Hq);
  float* tmp    = take((size_t)TOK*Hq);
  float* qb     = take((size_t)TOK*Hq);
  float* kb     = take((size_t)TOK*Hq);
  float* vb     = take((size_t)TOK*Hq);
  float* ctx    = take((size_t)TOK*Hq);
  float* hb     = take((size_t)TOK*DFFq);
  float* logits = take((size_t)TOK*NLq);

  dim3 g768(Hq/64, TOK/128);     // (12, 32)
  dim3 g3072(DFFq/64, TOK/128);  // (48, 32)

  embed_kernel<<<TOK,256,0,stream>>>(input_ids, token_emb, pos_emb, tmp);
  add_ln_kernel<<<TOK,256,0,stream>>>(x, tmp, (const float*)nullptr, ln_emb_s, ln_emb_b);

  for(int l=0;l<Lq;++l){
    const size_t wofs = (size_t)l*Hq*Hq;
    gemm_kernel<<<g768,256,0,stream>>>(x, Wq+wofs, bqv+(size_t)l*Hq, qb, TOK, Hq, Hq, 0);
    gemm_kernel<<<g768,256,0,stream>>>(x, Wk+wofs, bkv+(size_t)l*Hq, kb, TOK, Hq, Hq, 0);
    gemm_kernel<<<g768,256,0,stream>>>(x, Wv+wofs, bvv+(size_t)l*Hq, vb, TOK, Hq, Hq, 0);
    attn_kernel<<<1536,64,98304,stream>>>(qb, kb, vb, amask, ctx);
    gemm_kernel<<<g768,256,0,stream>>>(ctx, Wo+wofs, bov+(size_t)l*Hq, tmp, TOK, Hq, Hq, 0);
    add_ln_kernel<<<TOK,256,0,stream>>>(x, x, tmp, ln1_s+(size_t)l*Hq, ln1_b+(size_t)l*Hq);
    gemm_kernel<<<g3072,256,0,stream>>>(x, W1+(size_t)l*Hq*DFFq, b1v+(size_t)l*DFFq, hb, TOK, DFFq, Hq, 1);
    gemm_kernel<<<g768,256,0,stream>>>(hb, W2+(size_t)l*DFFq*Hq, b2v+(size_t)l*Hq, tmp, TOK, Hq, DFFq, 0);
    add_ln_kernel<<<TOK,256,0,stream>>>(x, x, tmp, ln2_s+(size_t)l*Hq, ln2_b+(size_t)l*Hq);
  }
  cls_kernel<<<TOK/256,256,0,stream>>>(x, cls_W, cls_b, logits);
  crf_kernel<<<1,256,0,stream>>>(logits, labels, amask, startT, endT, trans, (float*)d_out);
}
